// DGC_GRU_41583873360304
// MI455X (gfx1250) — matrix operand order repaired and checked by reading, not compiled, not run on hardware
//
#include <hip/hip_runtime.h>
#include <hip/hip_bf16.h>

// ---------------------------------------------------------------------------
// DGC-GRU for MI455X (gfx1250, wave32).
// Recurrent GEMM h[47104,128] @ Whh^T[128,384] done with
// v_wmma_f32_16x16x32_f16 (f16 inputs, f32 accumulate); gates + graph part on
// VALU. State traffic (~36MB/step) is the bottleneck -> keep weights f16 in
// L2, h mirrored in f16 for the WMMA A operand.
// ---------------------------------------------------------------------------

typedef __attribute__((ext_vector_type(16))) _Float16 v16h;
typedef __attribute__((ext_vector_type(8)))  float    v8f;

union V16HU { v16h v; unsigned int u[8]; };

#define BB   256
#define NN   184
#define RR   (BB * NN)        // 47104 rows
#define HH   128
#define TH   24
#define TF   24
#define TT_  (TH + TF)        // 48

// ---------------------------------------------------------------------------
// Fused GRU step: WMMA GEMM (gh = h @ Whh^T) -> LDS -> gates -> h update ->
// xn = h @ fcw^T. One block = 16 rows, 8 waves x 3 column tiles x 4 K-chunks.
// ---------------------------------------------------------------------------
__global__ __launch_bounds__(256)
void gru_step_kernel(const float* __restrict__ x, int inDim,
                     const _Float16* __restrict__ Whh16,   // [384,128] row-major (out,in)
                     const float* __restrict__ Wih,        // [384,inDim]
                     const float* __restrict__ bih,
                     const float* __restrict__ bhh,
                     const float* __restrict__ fcw,        // [128]
                     const float* __restrict__ fcb,        // [1]
                     float* __restrict__ h,                // [R,128] f32 state
                     _Float16* __restrict__ h16,           // [R,128] f16 mirror
                     float* __restrict__ xn,               // [R]
                     float* __restrict__ outp,             // preds or nullptr
                     int tstep)
{
    __shared__ float ghs[16][388];   // gh tiles (16 rows x 384 cols, padded)
    __shared__ float hns[16][132];   // new h for the xn reduction

    const int tid    = threadIdx.x;
    const int lane   = tid & 31;
    const int wave   = tid >> 5;
    const int rowBase = blockIdx.x * 16;
    const int hiHalf = (lane >= 16) ? 1 : 0;
    const int mrow   = rowBase + (lane & 15);      // A-matrix row (M) per lane

    // ---- load A: h16 rows, 4 K-chunks of 32 (ISA 16-bit A 16x32 layout) ----
    V16HU a[4];
#pragma unroll
    for (int kc = 0; kc < 4; ++kc) {
        const int kBase = kc * 32;
#pragma unroll
        for (int v = 0; v < 8; ++v) {
            const int koff = 2 * v + ((v >= 4) ? 8 : 0) + (hiHalf ? 8 : 0);
            a[kc].u[v] = *(const unsigned int*)(h16 + (size_t)mrow * HH + kBase + koff);
        }
    }

    // ---- 3 output column tiles per wave, accumulate over K=128 ----
#pragma unroll
    for (int j = 0; j < 3; ++j) {
        const int colBase = (wave * 3 + j) * 16;
        const int col     = colBase + (lane & 15); // B-matrix N per lane
        v8f acc = {0.f, 0.f, 0.f, 0.f, 0.f, 0.f, 0.f, 0.f};
#pragma unroll
        for (int kc = 0; kc < 4; ++kc) {
            V16HU b;
            const int kB = kc * 32 + (hiHalf ? 8 : 0);
#pragma unroll
            for (int v = 0; v < 8; ++v) {
                // B[k][col] = Whh[col][k]; pairs (2v,2v+1) contiguous in k
                b.u[v] = *(const unsigned int*)(Whh16 + (size_t)col * HH + kB + 2 * v + ((v >= 4) ? 8 : 0));
            }
            acc = __builtin_amdgcn_wmma_f32_16x16x32_f16(
                      false, a[kc].v, false, b.v, (short)0, acc, false, false);
        }
        // C/D layout: VGPR g holds M=g (lanes 0-15) / M=g+8 (lanes 16-31)
#pragma unroll
        for (int g = 0; g < 8; ++g)
            ghs[g + (hiHalf ? 8 : 0)][col] = acc[g];
    }
    __syncthreads();

    // ---- gates: gi on VALU (tiny in-dim), gh from LDS, update h ----
    for (int idx = tid; idx < 16 * HH; idx += 256) {
        const int row  = idx >> 7;
        const int c    = idx & 127;
        const int grow = rowBase + row;
        const float* xr = x + (size_t)grow * inDim;

        float gir = bih[c], giz = bih[c + 128], gin = bih[c + 256];
        for (int k = 0; k < inDim; ++k) {
            const float xv = xr[k];
            gir += xv * Wih[(size_t)c * inDim + k];
            giz += xv * Wih[(size_t)(c + 128) * inDim + k];
            gin += xv * Wih[(size_t)(c + 256) * inDim + k];
        }
        const float hr  = ghs[row][c]       + bhh[c];
        const float hz  = ghs[row][c + 128] + bhh[c + 128];
        const float hnn = ghs[row][c + 256] + bhh[c + 256];

        const float r = 1.f / (1.f + __expf(-(gir + hr)));
        const float z = 1.f / (1.f + __expf(-(giz + hz)));
        const float n = tanhf(gin + r * hnn);

        const float hold = h[(size_t)grow * HH + c];
        const float hnew = (1.f - z) * n + z * hold;
        h[(size_t)grow * HH + c]   = hnew;
        h16[(size_t)grow * HH + c] = (_Float16)hnew;
        hns[row][c] = hnew;
    }
    __syncthreads();

    // ---- xn = h_new @ fcw^T + fcb (one thread per row) ----
    if (tid < 16) {
        const int grow = rowBase + tid;
        float acc = fcb[0];
        for (int c = 0; c < HH; ++c) acc += hns[tid][c] * fcw[c];
        xn[grow] = acc;
        if (outp) {
            const int b = grow / NN, nidx = grow % NN;
            outp[((size_t)b * TF + tstep) * NN + nidx] = acc;
        }
    }
}

// ---------------------------------------------------------------------------
// Wind-gated degree: one wave per (b,i); deg_i = sum_j mask(i->j)
// ---------------------------------------------------------------------------
__global__ __launch_bounds__(256)
void wind_deg_kernel(const float* __restrict__ feature,
                     const float* __restrict__ adj,
                     const float* __restrict__ comp1,
                     const float* __restrict__ comp2,
                     float* __restrict__ dinv,
                     float* __restrict__ selfc, int tt)
{
    const int lane = threadIdx.x & 31;
    const int wave = threadIdx.x >> 5;
    const int gidx = blockIdx.x * 8 + wave;
    if (gidx >= RR) return;
    const int b = gidx / NN, i = gidx % NN;
    const float* f = feature + (((size_t)b * TT_ + tt) * NN + i) * 8;
    const float u = f[0], v = f[1];
    int cnt = 0;
    for (int j = lane; j < NN; j += 32) {
        const float w = u * comp1[i * NN + j] + v * comp2[i * NN + j];
        if (adj[i * NN + j] > 0.f && w >= 0.5f) cnt++;
    }
    for (int o = 16; o > 0; o >>= 1) cnt += __shfl_xor(cnt, o, 32);
    if (lane == 0) {
        const bool has = cnt > 0;
        dinv[gidx]  = has ? rsqrtf((float)cnt) : 0.f;
        selfc[gidx] = has ? 0.f : -1.f;
    }
}

// ---------------------------------------------------------------------------
// ChebConv aggregation: y[b,j,c] = -dinv_j * sum_i dinv_i*M[b,i,j]*x[b,i,c]
//                                  + selfc_j * x[b,j,c]; one wave per (b,j)
// ---------------------------------------------------------------------------
__global__ __launch_bounds__(256)
void cheb_y_kernel(const float* __restrict__ feature,
                   const float* __restrict__ adj,
                   const float* __restrict__ comp1,
                   const float* __restrict__ comp2,
                   const float* __restrict__ dinv,
                   const float* __restrict__ selfc,
                   const float* __restrict__ x,   // [R,9]
                   float* __restrict__ y,          // [R,9]
                   int tt)
{
    const int lane = threadIdx.x & 31;
    const int wave = threadIdx.x >> 5;
    const int gidx = blockIdx.x * 8 + wave;
    if (gidx >= RR) return;
    const int b = gidx / NN, j = gidx % NN;

    float acc[9];
#pragma unroll
    for (int c = 0; c < 9; ++c) acc[c] = 0.f;

    for (int i = lane; i < NN; i += 32) {
        const float a = adj[i * NN + j];
        const float* f = feature + (((size_t)b * TT_ + tt) * NN + i) * 8;
        const float w = f[0] * comp1[i * NN + j] + f[1] * comp2[i * NN + j];
        if (a > 0.f && w >= 0.5f) {
            const float di = dinv[b * NN + i];
            const float* xr = x + ((size_t)b * NN + i) * 9;
#pragma unroll
            for (int c = 0; c < 9; ++c) acc[c] += di * xr[c];
        }
    }
    for (int o = 16; o > 0; o >>= 1) {
#pragma unroll
        for (int c = 0; c < 9; ++c) acc[c] += __shfl_xor(acc[c], o, 32);
    }
    if (lane == 0) {
        const float dj = dinv[gidx], sc = selfc[gidx];
        const float* xj = x + (size_t)gidx * 9;
#pragma unroll
        for (int c = 0; c < 9; ++c)
            y[(size_t)gidx * 9 + c] = -dj * acc[c] + sc * xj[c];
    }
}

// g = sigmoid(x@w0 + y@w1 + cb); x2 = concat(x, g)  -> [R,10]
__global__ __launch_bounds__(256)
void g_x2_kernel(const float* __restrict__ x, const float* __restrict__ y,
                 const float* __restrict__ w0, const float* __restrict__ w1,
                 const float* __restrict__ cb, float* __restrict__ x2)
{
    const int row = blockIdx.x * 256 + threadIdx.x;
    if (row >= RR) return;
    float s = cb[0];
#pragma unroll
    for (int c = 0; c < 9; ++c) {
        const float xv = x[(size_t)row * 9 + c];
        s += xv * w0[c] + y[(size_t)row * 9 + c] * w1[c];
        x2[(size_t)row * 10 + c] = xv;
    }
    x2[(size_t)row * 10 + 9] = 1.f / (1.f + __expf(-s));
}

// forecast x = concat(xn, feature[:,tt])  -> [R,9]
__global__ __launch_bounds__(256)
void build_x_kernel(const float* __restrict__ xn,
                    const float* __restrict__ feature,
                    float* __restrict__ x, int tt)
{
    const int row = blockIdx.x * 256 + threadIdx.x;
    if (row >= RR) return;
    const int b = row / NN, n = row % NN;
    const float* f = feature + (((size_t)b * TT_ + tt) * NN + n) * 8;
    x[(size_t)row * 9 + 0] = xn[row];
#pragma unroll
    for (int c = 0; c < 8; ++c) x[(size_t)row * 9 + 1 + c] = f[c];
}

// history x = concat(xn, pm25_hist[:,t]) -> [R,2]
__global__ __launch_bounds__(256)
void build_xh_kernel(const float* __restrict__ xn,
                     const float* __restrict__ pm,
                     float* __restrict__ xh, int t)
{
    const int row = blockIdx.x * 256 + threadIdx.x;
    if (row >= RR) return;
    const int b = row / NN, n = row % NN;
    xh[(size_t)row * 2 + 0] = xn[row];
    xh[(size_t)row * 2 + 1] = pm[((size_t)b * TH + t) * NN + n];
}

// zero h / h16 / xn
__global__ __launch_bounds__(256)
void init_kernel(float* __restrict__ h, _Float16* __restrict__ h16,
                 float* __restrict__ xn)
{
    const size_t i = (size_t)blockIdx.x * 256 + threadIdx.x;
    if (i < (size_t)RR * HH) { h[i] = 0.f; h16[i] = (_Float16)0.f; }
    if (i < RR) xn[i] = 0.f;
}

// comp1/comp2 from angles + f16 weight conversion
__global__ __launch_bounds__(256)
void prep_kernel(const float* __restrict__ angles,
                 float* __restrict__ comp1, float* __restrict__ comp2,
                 const float* __restrict__ WhhH, const float* __restrict__ Whh,
                 _Float16* __restrict__ WhhH16, _Float16* __restrict__ Whh16)
{
    const int i = blockIdx.x * 256 + threadIdx.x;
    if (i < NN * NN) {
        const float a = angles[i];
        comp1[i] = cosf(a);
        comp2[i] = cosf(a - 1.57079632679489662f);
    }
    if (i < 384 * HH) {
        WhhH16[i] = (_Float16)WhhH[i];
        Whh16[i]  = (_Float16)Whh[i];
    }
}

// ---------------------------------------------------------------------------
extern "C" void kernel_launch(void* const* d_in, const int* in_sizes, int n_in,
                              void* d_out, int out_size, void* d_ws, size_t ws_size,
                              hipStream_t stream) {
    const float* feature   = (const float*)d_in[0];
    const float* pm25_hist = (const float*)d_in[1];
    const float* adj_mat   = (const float*)d_in[2];
    const float* angles    = (const float*)d_in[3];
    const float* W_ih_hist = (const float*)d_in[4];
    const float* W_hh_hist = (const float*)d_in[5];
    const float* b_ih_hist = (const float*)d_in[6];
    const float* b_hh_hist = (const float*)d_in[7];
    const float* fc_hist_w = (const float*)d_in[8];
    const float* fc_hist_b = (const float*)d_in[9];
    const float* cheb_w0   = (const float*)d_in[10];
    const float* cheb_w1   = (const float*)d_in[11];
    const float* cheb_b    = (const float*)d_in[12];
    const float* W_ih      = (const float*)d_in[13];
    const float* W_hh      = (const float*)d_in[14];
    const float* b_ih      = (const float*)d_in[15];
    const float* b_hh      = (const float*)d_in[16];
    const float* fc_out_w  = (const float*)d_in[17];
    const float* fc_out_b  = (const float*)d_in[18];
    float* out = (float*)d_out;

    // ---- workspace carve (all offsets 256B-aligned) ----
    char* base = (char*)d_ws;
    size_t off = 0;
    auto carve = [&](size_t bytes) {
        void* p = base + off;
        off += (bytes + 255) & ~(size_t)255;
        return p;
    };
    float*    h       = (float*)   carve((size_t)RR * HH * 4);
    _Float16* h16     = (_Float16*)carve((size_t)RR * HH * 2);
    float*    xn      = (float*)   carve((size_t)RR * 4);
    float*    x       = (float*)   carve((size_t)RR * 9 * 4);
    float*    x2      = (float*)   carve((size_t)RR * 10 * 4);
    float*    xh      = (float*)   carve((size_t)RR * 2 * 4);
    float*    dinv    = (float*)   carve((size_t)RR * 4);
    float*    selfc   = (float*)   carve((size_t)RR * 4);
    float*    y       = (float*)   carve((size_t)RR * 9 * 4);
    float*    comp1   = (float*)   carve((size_t)NN * NN * 4);
    float*    comp2   = (float*)   carve((size_t)NN * NN * 4);
    _Float16* WhhH16  = (_Float16*)carve((size_t)384 * HH * 2);
    _Float16* Whh16   = (_Float16*)carve((size_t)384 * HH * 2);
    (void)ws_size; (void)in_sizes; (void)n_in; (void)out_size;

    const int rowsGrid = (RR + 255) / 256;                 // 184 blocks
    const int gruGrid  = RR / 16;                          // 2944 blocks
    const int waveGrid = (RR + 7) / 8;                     // 5888 blocks
    const int initGrid = (int)(((size_t)RR * HH + 255) / 256);

    init_kernel<<<initGrid, 256, 0, stream>>>(h, h16, xn);
    prep_kernel<<<(384 * HH + 255) / 256, 256, 0, stream>>>(
        angles, comp1, comp2, W_hh_hist, W_hh, WhhH16, Whh16);

    // ---- history GRU (in-dim 2) ----
    for (int t = 0; t < TH; ++t) {
        build_xh_kernel<<<rowsGrid, 256, 0, stream>>>(xn, pm25_hist, xh, t);
        gru_step_kernel<<<gruGrid, 256, 0, stream>>>(
            xh, 2, WhhH16, W_ih_hist, b_ih_hist, b_hh_hist,
            fc_hist_w, fc_hist_b, h, h16, xn, nullptr, 0);
    }

    // ---- forecast: graph gating + ChebConv + GRU (in-dim 10) ----
    for (int t = 0; t < TF; ++t) {
        const int tt = TH + t;
        build_x_kernel<<<rowsGrid, 256, 0, stream>>>(xn, feature, x, tt);
        wind_deg_kernel<<<waveGrid, 256, 0, stream>>>(
            feature, adj_mat, comp1, comp2, dinv, selfc, tt);
        cheb_y_kernel<<<waveGrid, 256, 0, stream>>>(
            feature, adj_mat, comp1, comp2, dinv, selfc, x, y, tt);
        g_x2_kernel<<<rowsGrid, 256, 0, stream>>>(x, y, cheb_w0, cheb_w1, cheb_b, x2);
        gru_step_kernel<<<gruGrid, 256, 0, stream>>>(
            x2, 10, Whh16, W_ih, b_ih, b_hh,
            fc_out_w, fc_out_b, h, h16, xn, out, t);
    }
}
